// TransformerBlock_77876347011240
// MI455X (gfx1250) — matrix operand order repaired and checked by reading, not compiled, not run on hardware
//
#include <hip/hip_runtime.h>
#include <hip/hip_bf16.h>
#include <math.h>

typedef _Float16 v16h __attribute__((ext_vector_type(16)));
typedef _Float16 h8   __attribute__((ext_vector_type(8)));
typedef _Float16 h4   __attribute__((ext_vector_type(4)));
typedef float    v8f  __attribute__((ext_vector_type(8)));

#define NP  32768
#define NPL 32768ull
#define CSTR 40   // LDS row stride in halfs (16B-aligned rows, conflict-free)

// ---------------------------------------------------------------------------
// Weight packing: f32 -> f16 plain copy (linear weights, row-major (O,K))
// ---------------------------------------------------------------------------
__global__ void pack_linear_k(const float* __restrict__ src, _Float16* __restrict__ dst, int n) {
    int i = blockIdx.x * 256 + threadIdx.x;
    if (i < n) dst[i] = (_Float16)src[i];
}

// Conv weight pack: w (64, 64, taps) f32 -> Bp[(tap*2+ch)][co][kc] f16
// (K ordering inside the implicit GEMM is k = tap*64 + cin, split into 32-wide steps)
__global__ void pack_conv_k(const float* __restrict__ w, _Float16* __restrict__ dst,
                            int taps, int ntot) {
    int i = blockIdx.x * 256 + threadIdx.x;
    if (i >= ntot) return;
    int kc = i & 31;
    int co = (i >> 5) & 63;
    int s  = i >> 11;
    int tap = s >> 1, ch = s & 1;
    int cin = ch * 32 + kc;
    dst[i] = (_Float16)w[(size_t)(co * 64 + cin) * taps + tap];
}

// ---------------------------------------------------------------------------
// LayerNorm over C=64, x is (C,N) f32 -> ln16 (N,64) f16 rows
// ---------------------------------------------------------------------------
__global__ void __launch_bounds__(256) ln_k(const float* __restrict__ x,
                                            const float* __restrict__ g,
                                            const float* __restrict__ b,
                                            _Float16* __restrict__ out) {
    int n = blockIdx.x * 256 + threadIdx.x;
    float v[64];
    float m = 0.f;
#pragma unroll
    for (int c = 0; c < 64; ++c) { v[c] = x[(size_t)c * NPL + n]; m += v[c]; }
    m *= (1.f / 64.f);
    float var = 0.f;
#pragma unroll
    for (int c = 0; c < 64; ++c) { float d = v[c] - m; var += d * d; }
    var *= (1.f / 64.f);
    float rs = rsqrtf(var + 1e-5f);
#pragma unroll
    for (int c = 0; c < 64; ++c)
        out[(size_t)n * 64 + c] = (_Float16)((v[c] - m) * rs * g[c] + b[c]);
}

// ---------------------------------------------------------------------------
// LDS-less WMMA GEMM: out(OF, npos) = Wp(OF,64) * X(npos,64)^T (+bias)
// wave tile: 16 (out-feat, WMMA-M) x 16 (positions, WMMA-N), K=64 = 2 wmma.
// ---------------------------------------------------------------------------
template <bool HAS_BIAS>
__global__ void __launch_bounds__(256) gemm16_k(const _Float16* __restrict__ Wp,
                                                const _Float16* __restrict__ X,
                                                const float* __restrict__ bias,
                                                float* __restrict__ out, int npos) {
    const int lane = threadIdx.x & 31;
    const int wave = threadIdx.x >> 5;
    const int lh   = lane >> 4;      // half-wave
    const int l16  = lane & 15;
    const int n0   = blockIdx.x * 128 + wave * 16;
    const int co0  = blockIdx.y * 16;
    const _Float16* wrow = Wp + (size_t)(co0 + l16) * 64;
    const _Float16* xrow = X  + (size_t)(n0 + l16) * 64;
    v8f acc = {};
#pragma unroll
    for (int ks = 0; ks < 2; ++ks) {
        // A (16-bit 16x32): lane half selects K base {0,8} + {16,24}
        h8 alo = *(const h8*)(wrow + ks * 32 + lh * 8);
        h8 ahi = *(const h8*)(wrow + ks * 32 + lh * 8 + 16);
        // B (16-bit 32x16): lane half selects K base {0,8} + {16,24}
        h8 blo = *(const h8*)(xrow + ks * 32 + lh * 8);
        h8 bhi = *(const h8*)(xrow + ks * 32 + lh * 8 + 16);
        v16h A, B;
#pragma unroll
        for (int i = 0; i < 8; ++i) { A[i] = alo[i]; A[8 + i] = ahi[i];
                                      B[i] = blo[i]; B[8 + i] = bhi[i]; }
        acc = __builtin_amdgcn_wmma_f32_16x16x32_f16(false, A, false, B,
                                                     (short)0, acc, false, false);
    }
    float bv[8];
#pragma unroll
    for (int r = 0; r < 8; ++r) bv[r] = 0.f;
    if (HAS_BIAS) {
        // bias rows co0 + 8*lh + {0..7} are contiguous: two aligned float4 loads
        const float4* bp = (const float4*)(bias + co0 + 8 * lh);
        float4 b0 = bp[0], b1 = bp[1];
        bv[0] = b0.x; bv[1] = b0.y; bv[2] = b0.z; bv[3] = b0.w;
        bv[4] = b1.x; bv[5] = b1.y; bv[6] = b1.z; bv[7] = b1.w;
    }
#pragma unroll
    for (int r = 0; r < 8; ++r) {
        int co = co0 + r + 8 * lh;              // C layout: M = r (+8 for hi half)
        out[(size_t)co * npos + (n0 + l16)] = acc[r] + bv[r];
    }
}

// ---------------------------------------------------------------------------
// Implicit-GEMM conv (3^3 or 1^1), C=64->64 over 32^3, pad ks/2.
// in16 (64, N) f16 ; Bp packed [NTAPS*2][64][32] f16 ; out (64,N) f32.
// HAS_RES -> out = conv + bias + res   (used for the final 1x1 conv)
// ---------------------------------------------------------------------------
template <int NTAPS, bool HAS_RES>
__global__ void __launch_bounds__(256) conv_wmma_k(const _Float16* __restrict__ in16,
                                                   const _Float16* __restrict__ Bp,
                                                   const float* __restrict__ bias,
                                                   const float* __restrict__ res,
                                                   float* __restrict__ out) {
    __shared__ alignas(16) _Float16 sI[2][32 * CSTR]; // input tile  [m][k]
    __shared__ alignas(16) _Float16 sW[2][64 * CSTR]; // weight tile [co][k]
    const int t = threadIdx.x;
    const int a = blockIdx.x >> 5;
    const int b = blockIdx.x & 31;
    const int lane = t & 31, wave = t >> 5;
    const int lh = lane >> 4, l16 = lane & 15;
    const int wco = wave >> 1, wm = wave & 1;   // 4 co-tiles x 2 m-tiles
    constexpr int STEPS = NTAPS * 2;
    // loader mapping
    const int lm = t >> 3;          // spatial m 0..31
    const int lk = (t & 7) * 4;     // cin sub   0..28
    const int wc = t >> 2;          // co        0..63
    const int wk = (t & 3) * 8;     // k sub     0,8,16,24
    v8f acc = {};

    auto load_step = [&](int s, int buf) {
        const int tap = s >> 1, ch = s & 1;
        int dz = 0, dy = 0, dx = 0;
        if (NTAPS == 27) { dz = tap / 9 - 1; dy = (tap / 3) % 3 - 1; dx = tap % 3 - 1; }
        int z = a + dz, y = b + dy, xx = lm + dx;
        bool ok = ((unsigned)z < 32u) & ((unsigned)y < 32u) & ((unsigned)xx < 32u);
        h4 v = {};
        if (ok) {
            const _Float16* p = in16 + (size_t)(ch * 32 + lk) * NPL + ((z * 32 + y) * 32 + xx);
#pragma unroll
            for (int i = 0; i < 4; ++i) v[i] = p[(size_t)i * NPL];
        }
        *(h4*)(&sI[buf][lm * CSTR + lk]) = v;
        *(h8*)(&sW[buf][wc * CSTR + wk]) =
            *(const h8*)(Bp + ((size_t)s * 64 + wc) * 32 + wk);
    };

    load_step(0, 0);
#pragma unroll 2
    for (int s = 0; s < STEPS; ++s) {
        __syncthreads();
        if (s + 1 < STEPS) load_step(s + 1, (s + 1) & 1);
        const int buf = s & 1;
        const _Float16* wr = &sW[buf][(wco * 16 + l16) * CSTR + lh * 8];
        h8 alo = *(const h8*)wr;
        h8 ahi = *(const h8*)(wr + 16);
        const _Float16* ir = &sI[buf][(wm * 16 + l16) * CSTR + lh * 8];
        h8 blo = *(const h8*)ir;
        h8 bhi = *(const h8*)(ir + 16);
        v16h A, B;
#pragma unroll
        for (int i = 0; i < 8; ++i) { A[i] = alo[i]; A[8 + i] = ahi[i];
                                      B[i] = blo[i]; B[8 + i] = bhi[i]; }
        acc = __builtin_amdgcn_wmma_f32_16x16x32_f16(false, A, false, B,
                                                     (short)0, acc, false, false);
    }
    const int n = (a * 32 + b) * 32 + wm * 16 + l16;
    // bias rows wco*16 + 8*lh + {0..7} contiguous: two float4 loads
    const float4* bp = (const float4*)(bias + wco * 16 + 8 * lh);
    float4 b0 = bp[0], b1 = bp[1];
    float bv[8] = { b0.x, b0.y, b0.z, b0.w, b1.x, b1.y, b1.z, b1.w };
#pragma unroll
    for (int r = 0; r < 8; ++r) {
        int co = wco * 16 + r + 8 * lh;
        size_t idx = (size_t)co * NPL + n;
        float v = acc[r] + bv[r];
        if (HAS_RES) v += res[idx];
        out[idx] = v;
    }
}

// ---------------------------------------------------------------------------
// qns = (l2norm(q) + qemb) * softplus(temp2) ; q rows 0..63 of Yt (feat, N)
// ---------------------------------------------------------------------------
__global__ void __launch_bounds__(256) qns_k(const float* __restrict__ Yt,
                                             const float* __restrict__ temp2,
                                             const float* __restrict__ qemb,
                                             float* __restrict__ qns) {
    int id = blockIdx.x * 256 + threadIdx.x;
    int h = id >> 15, n = id & (NP - 1);
    float q[16]; float ss = 0.f;
#pragma unroll
    for (int e = 0; e < 16; ++e) { q[e] = Yt[(size_t)(h * 16 + e) * NPL + n]; ss += q[e] * q[e]; }
    float inv = 1.f / fmaxf(sqrtf(ss), 1e-12f);
    float t2 = temp2[h];
    float sp = (t2 > 0.f) ? (t2 + log1pf(expf(-t2))) : log1pf(expf(t2));
#pragma unroll
    for (int e = 0; e < 16; ++e)
        qns[(size_t)(h * 16 + e) * NPL + n] = (q[e] * inv + qemb[h * 16 + e]) * sp;
}

// ---------------------------------------------------------------------------
// Spatial window attention; replicates the reference's reshape quirk on x_sa.
// ---------------------------------------------------------------------------
__global__ void __launch_bounds__(256) spatial_k(const float* __restrict__ qns,
                                                 const float* __restrict__ Yt,
                                                 const float* __restrict__ rpb,
                                                 _Float16* __restrict__ xsa16) {
    int id = blockIdx.x * 256 + threadIdx.x;
    int h = id >> 15, n = id & (NP - 1);
    int z = n >> 10, y = (n >> 5) & 31, xx = n & 31;
    float qv[16];
#pragma unroll
    for (int e = 0; e < 16; ++e) qv[e] = qns[(size_t)(h * 16 + e) * NPL + n];
    const float* kb = Yt + (size_t)(64 + h * 16) * NPL;   // k rows
    const float* vb = Yt + (size_t)(192 + h * 16) * NPL;  // v_sa rows
    float lg[27];
    float mx = -3.0e38f;
#pragma unroll
    for (int tap = 0; tap < 27; ++tap) {
        const int di = tap / 9 - 1, dj = (tap / 3) % 3 - 1, dk = tap % 3 - 1;
        int zz = z + di, yy = y + dj, xk = xx + dk;
        bool ok = ((unsigned)zz < 32u) && ((unsigned)yy < 32u) && ((unsigned)xk < 32u);
        float l = -3.0e38f;
        if (ok) {
            int nb = (zz * 32 + yy) * 32 + xk;
            float s = rpb[h * 27 + tap];
#pragma unroll
            for (int e = 0; e < 16; ++e) s += qv[e] * kb[(size_t)e * NPL + nb];
            l = s;
        }
        lg[tap] = l;
        mx = fmaxf(mx, l);
    }
    float sum = 0.f;
#pragma unroll
    for (int tap = 0; tap < 27; ++tap) {
        float p = (lg[tap] <= -1.0e38f) ? 0.f : expf(lg[tap] - mx);
        lg[tap] = p; sum += p;
    }
    float inv = 1.f / sum;
    float o[16];
#pragma unroll
    for (int e = 0; e < 16; ++e) o[e] = 0.f;
#pragma unroll
    for (int tap = 0; tap < 27; ++tap) {
        const int di = tap / 9 - 1, dj = (tap / 3) % 3 - 1, dk = tap % 3 - 1;
        int zz = z + di, yy = y + dj, xk = xx + dk;
        if (((unsigned)zz < 32u) && ((unsigned)yy < 32u) && ((unsigned)xk < 32u)) {
            int nb = (zz * 32 + yy) * 32 + xk;
            float c = lg[tap] * inv;
#pragma unroll
            for (int e = 0; e < 16; ++e) o[e] += c * vb[(size_t)e * NPL + nb];
        }
    }
    // reshape quirk: (NH,N,hd) flattened straight to (N, C)
    size_t row = (size_t)(h * 8192 + (n >> 2));
    int cb = (n & 3) * 16;
#pragma unroll
    for (int e = 0; e < 16; ++e) xsa16[row * 64 + cb + e] = (_Float16)o[e];
}

// ---------------------------------------------------------------------------
// Channel attention, stage 1: per-(head,chunk) partial Gram + norms (no atomics)
// partial layout per (h,chunk): [0..255]=S(d,e), [256..271]=sum q^2, [272..287]=sum k^2
// ---------------------------------------------------------------------------
__global__ void __launch_bounds__(256) chanpart_k(const float* __restrict__ Yt,
                                                  float* __restrict__ partial) {
    int h = blockIdx.y, chunk = blockIdx.x;
    int d = threadIdx.x >> 4, e = threadIdx.x & 15;
    const float* qrow = Yt + (size_t)(h * 16 + d) * NPL;
    const float* krow = Yt + (size_t)(64 + h * 16 + e) * NPL;
    float s = 0.f, sq = 0.f, sk = 0.f;
    int n0 = chunk * 1024;
    for (int i = 0; i < 1024; ++i) {
        float qv = qrow[n0 + i], kv = krow[n0 + i];
        s += qv * kv; sq += qv * qv; sk += kv * kv;
    }
    float* P = partial + ((size_t)h * 32 + chunk) * 288;
    P[d * 16 + e] = s;
    if (e == 0) P[256 + d] = sq;
    if (d == 0) P[272 + e] = sk;
}

// stage 2: reduce chunks, normalize (cosine), scale by temp, row softmax -> Aca
__global__ void __launch_bounds__(256) chanfinal_k(const float* __restrict__ partial,
                                                   const float* __restrict__ temp,
                                                   float* __restrict__ Aca) {
    int h = blockIdx.x;
    int d = threadIdx.x >> 4, e = threadIdx.x & 15;
    __shared__ float S[256], NQ[16], NK[16], R[256];
    float s = 0.f;
    for (int c = 0; c < 32; ++c) s += partial[((size_t)h * 32 + c) * 288 + d * 16 + e];
    S[d * 16 + e] = s;
    if (e == 0) { float q = 0.f; for (int c = 0; c < 32; ++c) q += partial[((size_t)h * 32 + c) * 288 + 256 + d];
                  NQ[d] = fmaxf(sqrtf(q), 1e-12f); }
    if (d == 0) { float k = 0.f; for (int c = 0; c < 32; ++c) k += partial[((size_t)h * 32 + c) * 288 + 272 + e];
                  NK[e] = fmaxf(sqrtf(k), 1e-12f); }
    __syncthreads();
    R[d * 16 + e] = S[d * 16 + e] / (NQ[d] * NK[e]) * temp[h];
    __syncthreads();
    if (e == 0) {
        float mx = -3.0e38f;
        for (int j = 0; j < 16; ++j) mx = fmaxf(mx, R[d * 16 + j]);
        float ex[16]; float sum = 0.f;
        for (int j = 0; j < 16; ++j) { ex[j] = expf(R[d * 16 + j] - mx); sum += ex[j]; }
        float inv = 1.f / sum;
        for (int j = 0; j < 16; ++j) Aca[(size_t)(h * 16 + d) * 16 + j] = ex[j] * inv;
    }
}

// x_ca[n][h*16+d] = sum_e Aca[h][d][e] * v_ca[h][e][n]  -> f16 (N,64) rows
__global__ void __launch_bounds__(256) xca_k(const float* __restrict__ Yt,
                                             const float* __restrict__ Aca,
                                             _Float16* __restrict__ xca16) {
    __shared__ float sA[1024];
    for (int i = threadIdx.x; i < 1024; i += 256) sA[i] = Aca[i];
    __syncthreads();
    int n = blockIdx.x * 256 + threadIdx.x;
#pragma unroll
    for (int h = 0; h < 4; ++h) {
        float v[16];
#pragma unroll
        for (int e = 0; e < 16; ++e) v[e] = Yt[(size_t)(128 + h * 16 + e) * NPL + n];
#pragma unroll
        for (int d = 0; d < 16; ++d) {
            float s = 0.f;
#pragma unroll
            for (int e = 0; e < 16; ++e) s += sA[(h * 16 + d) * 16 + e] * v[e];
            xca16[(size_t)n * 64 + h * 16 + d] = (_Float16)s;
        }
    }
}

// skip = x + gamma * epa   (f32 (C,N) + f16 mirror for the conv)
__global__ void __launch_bounds__(256) skip_k(const float* __restrict__ x,
                                              const float* __restrict__ gamma,
                                              const float* __restrict__ epa,
                                              float* __restrict__ skip,
                                              _Float16* __restrict__ skip16) {
    size_t idx = (size_t)blockIdx.x * 256 + threadIdx.x;
    int c = (int)(idx >> 15);
    float s = x[idx] + gamma[c] * epa[idx];
    skip[idx] = s;
    skip16[idx] = (_Float16)s;
}

// BatchNorm stats per channel (biased var), deterministic tree reduction
__global__ void __launch_bounds__(256) bnstats_k(const float* __restrict__ x,
                                                 float* __restrict__ mv) {
    __shared__ float rs[256], rs2[256];
    int c = blockIdx.x, t = threadIdx.x;
    const float* p = x + (size_t)c * NPL;
    float s = 0.f, s2 = 0.f;
    for (int i = t; i < NP; i += 256) { float v = p[i]; s += v; s2 += v * v; }
    rs[t] = s; rs2[t] = s2; __syncthreads();
    for (int k = 128; k > 0; k >>= 1) {
        if (t < k) { rs[t] += rs[t + k]; rs2[t] += rs2[t + k]; }
        __syncthreads();
    }
    if (t == 0) { float m = rs[0] / NP; mv[c] = m; mv[64 + c] = rs2[0] / NP - m * m; }
}

// y = lrelu( bn(x) [+ res] ) -> f32 + f16 mirror
template <bool HAS_RES>
__global__ void __launch_bounds__(256) bnapply_k(const float* __restrict__ x,
                                                 const float* __restrict__ mv,
                                                 const float* __restrict__ g,
                                                 const float* __restrict__ be,
                                                 const float* __restrict__ res,
                                                 float* __restrict__ y,
                                                 _Float16* __restrict__ y16) {
    size_t idx = (size_t)blockIdx.x * 256 + threadIdx.x;
    int c = (int)(idx >> 15);
    float t = (x[idx] - mv[c]) * rsqrtf(mv[64 + c] + 1e-5f) * g[c] + be[c];
    if (HAS_RES) t += res[idx];
    t = (t > 0.f) ? t : 0.01f * t;
    y[idx] = t;
    y16[idx] = (_Float16)t;
}

// ---------------------------------------------------------------------------
extern "C" void kernel_launch(void* const* d_in, const int* in_sizes, int n_in,
                              void* d_out, int out_size, void* d_ws, size_t ws_size,
                              hipStream_t stream) {
    const float* x      = (const float*)d_in[0];
    const float* ln_g   = (const float*)d_in[1];
    const float* ln_b   = (const float*)d_in[2];
    const float* gamma  = (const float*)d_in[3];
    const float* qkvv_w = (const float*)d_in[4];
    const float* temp   = (const float*)d_in[5];
    const float* temp2  = (const float*)d_in[6];
    const float* rpb    = (const float*)d_in[7];
    const float* qemb   = (const float*)d_in[8];
    const float* op1_w  = (const float*)d_in[9];
    const float* op1_b  = (const float*)d_in[10];
    const float* op2_w  = (const float*)d_in[11];
    const float* op2_b  = (const float*)d_in[12];
    const float* cw[4]  = { (const float*)d_in[13], (const float*)d_in[17],
                            (const float*)d_in[21], (const float*)d_in[25] };
    const float* cb[4]  = { (const float*)d_in[14], (const float*)d_in[18],
                            (const float*)d_in[22], (const float*)d_in[26] };
    const float* cg[4]  = { (const float*)d_in[15], (const float*)d_in[19],
                            (const float*)d_in[23], (const float*)d_in[27] };
    const float* cbe[4] = { (const float*)d_in[16], (const float*)d_in[20],
                            (const float*)d_in[24], (const float*)d_in[28] };
    const float* c8_w   = (const float*)d_in[29];
    const float* c8_b   = (const float*)d_in[30];
    float* out = (float*)d_out;
    char*  W   = (char*)d_ws;

    size_t o = 0;
    auto take = [&](size_t bytes) { size_t r = o; o += (bytes + 255) & ~(size_t)255; return r; };
    const size_t o_qkvvp = take(256 * 64 * 2);
    const size_t o_op1p  = take(32 * 64 * 2);
    const size_t o_op2p  = take(32 * 64 * 2);
    const size_t o_c8p   = take(2 * 64 * 32 * 2);
    size_t o_cwp[4]; for (int i = 0; i < 4; ++i) o_cwp[i] = take(54 * 64 * 32 * 2);
    const size_t o_ln16  = take(NPL * 64 * 2);
    const size_t o_Yt    = take(256 * NPL * 4);
    const size_t o_qns   = take(64 * NPL * 4);
    const size_t o_xsa   = take(NPL * 64 * 2);
    const size_t o_xca   = take(NPL * 64 * 2);
    const size_t o_part  = take(4 * 32 * 288 * 4);
    const size_t o_Aca   = take(4 * 16 * 16 * 4);
    const size_t o_epa   = take(64 * NPL * 4);
    const size_t o_skip  = take(64 * NPL * 4);
    const size_t o_sk16  = take(NPL * 64 * 2);
    const size_t o_raw   = take(64 * NPL * 4);
    const size_t o_mv    = take(128 * 4);
    const size_t o_actA  = take(64 * NPL * 4);
    const size_t o_actA16= take(NPL * 64 * 2);
    const size_t o_actB  = take(64 * NPL * 4);
    const size_t o_actB16= take(NPL * 64 * 2);
    if (ws_size < o) return;  // workspace too small; bail deterministically

    _Float16* qkvvp = (_Float16*)(W + o_qkvvp);
    _Float16* op1p  = (_Float16*)(W + o_op1p);
    _Float16* op2p  = (_Float16*)(W + o_op2p);
    _Float16* c8p   = (_Float16*)(W + o_c8p);
    _Float16* cwp[4]; for (int i = 0; i < 4; ++i) cwp[i] = (_Float16*)(W + o_cwp[i]);
    _Float16* ln16  = (_Float16*)(W + o_ln16);
    float*    Yt    = (float*)(W + o_Yt);
    float*    qns   = (float*)(W + o_qns);
    _Float16* xsa16 = (_Float16*)(W + o_xsa);
    _Float16* xca16 = (_Float16*)(W + o_xca);
    float*    part  = (float*)(W + o_part);
    float*    Aca   = (float*)(W + o_Aca);
    float*    epa   = (float*)(W + o_epa);
    float*    skip  = (float*)(W + o_skip);
    _Float16* skip16= (_Float16*)(W + o_sk16);
    float*    raw   = (float*)(W + o_raw);
    float*    mv    = (float*)(W + o_mv);
    float*    actA  = (float*)(W + o_actA);
    _Float16* actA16= (_Float16*)(W + o_actA16);
    float*    actB  = (float*)(W + o_actB);
    _Float16* actB16= (_Float16*)(W + o_actB16);

    // --- weight packing (f32 -> f16, conv weights into per-step tile layout) ---
    pack_linear_k<<<64, 256, 0, stream>>>(qkvv_w, qkvvp, 256 * 64);
    pack_linear_k<<<8, 256, 0, stream>>>(op1_w, op1p, 32 * 64);
    pack_linear_k<<<8, 256, 0, stream>>>(op2_w, op2p, 32 * 64);
    for (int i = 0; i < 4; ++i)
        pack_conv_k<<<432, 256, 0, stream>>>(cw[i], cwp[i], 27, 54 * 64 * 32);
    pack_conv_k<<<16, 256, 0, stream>>>(c8_w, c8p, 1, 2 * 64 * 32);

    // --- attention ---
    ln_k<<<128, 256, 0, stream>>>(x, ln_g, ln_b, ln16);
    gemm16_k<false><<<dim3(256, 16), 256, 0, stream>>>(qkvvp, ln16, nullptr, Yt, NP);
    qns_k<<<512, 256, 0, stream>>>(Yt, temp2, qemb, qns);
    chanpart_k<<<dim3(32, 4), 256, 0, stream>>>(Yt, part);
    chanfinal_k<<<4, 256, 0, stream>>>(part, temp, Aca);
    spatial_k<<<512, 256, 0, stream>>>(qns, Yt, rpb, xsa16);
    xca_k<<<128, 256, 0, stream>>>(Yt, Aca, xca16);
    gemm16_k<true><<<dim3(256, 2), 256, 0, stream>>>(op1p, xsa16, op1_b, epa, NP);
    gemm16_k<true><<<dim3(256, 2), 256, 0, stream>>>(op2p, xca16, op2_b, epa + 32 * NPL, NP);
    skip_k<<<8192, 256, 0, stream>>>(x, gamma, epa, skip, skip16);

    // --- resblock 1 ---
    conv_wmma_k<27, false><<<1024, 256, 0, stream>>>(skip16, cwp[0], cb[0], nullptr, raw);
    bnstats_k<<<64, 256, 0, stream>>>(raw, mv);
    bnapply_k<false><<<8192, 256, 0, stream>>>(raw, mv, cg[0], cbe[0], nullptr, actA, actA16);
    conv_wmma_k<27, false><<<1024, 256, 0, stream>>>(actA16, cwp[1], cb[1], nullptr, raw);
    bnstats_k<<<64, 256, 0, stream>>>(raw, mv);
    bnapply_k<true><<<8192, 256, 0, stream>>>(raw, mv, cg[1], cbe[1], skip, actB, actB16);

    // --- resblock 2 ---
    conv_wmma_k<27, false><<<1024, 256, 0, stream>>>(actB16, cwp[2], cb[2], nullptr, raw);
    bnstats_k<<<64, 256, 0, stream>>>(raw, mv);
    bnapply_k<false><<<8192, 256, 0, stream>>>(raw, mv, cg[2], cbe[2], nullptr, actA, actA16);
    conv_wmma_k<27, false><<<1024, 256, 0, stream>>>(actA16, cwp[3], cb[3], nullptr, raw);
    bnstats_k<<<64, 256, 0, stream>>>(raw, mv);
    bnapply_k<true><<<8192, 256, 0, stream>>>(raw, mv, cg[3], cbe[3], actB, actA, actA16);

    // --- final 1x1x1 conv fused with skip residual, writes d_out ---
    conv_wmma_k<1, true><<<1024, 256, 0, stream>>>(actA16, c8p, c8_b, skip, out);
}
